// CVRP_Decoder_34935263986374
// MI455X (gfx1250) — compile-verified, matrix operand order checked by reading, not run on hardware
//
#include <hip/hip_runtime.h>

typedef _Float16 v16h __attribute__((ext_vector_type(16)));
typedef _Float16 h8   __attribute__((ext_vector_type(8)));
typedef float    v8f  __attribute__((ext_vector_type(8)));
typedef unsigned int u32x4 __attribute__((ext_vector_type(4)));
typedef int          i32x8 __attribute__((ext_vector_type(8)));
typedef int          i32x4 __attribute__((ext_vector_type(4)));

#if defined(__has_builtin)
#if __has_builtin(__builtin_amdgcn_tensor_load_to_lds)
#define HAVE_TDM 1
#endif
#endif
#ifndef HAVE_TDM
#define HAVE_TDM 0
#endif

namespace {
constexpr int cB = 64, cN = 256, cM = 1000, cS = 500, cE = 128, cH = 8, cD = 16, cHD = 128;
constexpr float NEG        = -1e9f;
constexpr float CLIPc      = 10.0f;
constexpr float INV_SQRT_E = 0.08838834764831845f;  // 1/sqrt(128)
constexpr float INV_SQRT_D = 0.25f;                 // 1/sqrt(16)
}

__device__ inline v8f zero_v8f() {
  v8f z;
#pragma unroll
  for (int i = 0; i < 8; ++i) z[i] = 0.0f;
  return z;
}
__device__ inline v16h zero_v16h() {
  v16h z;
#pragma unroll
  for (int i = 0; i < 16; ++i) z[i] = (_Float16)0.0f;
  return z;
}
__device__ inline v8f wmma_f16(v16h a, v16h b, v8f c) {
  // D = A(16x32 f16) * B(32x16 f16) + C(16x16 f32)
  return __builtin_amdgcn_wmma_f32_16x16x32_f16(false, a, false, b, (short)0, c,
                                                false, false);
}
__device__ inline void wait_dscnt0() {
  asm volatile("s_wait_dscnt 0x0" ::: "memory");
}

#if HAVE_TDM
// 2D tensor DMA: global (row-major, f32, row stride `stride0` elems) -> LDS
// tile of tile_d0 x tile_d1 elements; rows/cols beyond tensor_d0/tensor_d1
// (measured from the tile origin) are zero-filled by the TDM.
// amdgpu-toolchain (clang-23) signature: 6 args
//   (u32x4 g0, i32x8 g1, i32x4 g2, i32x4 g3, i32x8 g23hi, i32 cpol)
__device__ inline void tdm_load_2d(unsigned lds_off, unsigned long long ga,
                                   int tensor_d0, int tensor_d1, int tile_d0,
                                   int tile_d1, int stride0) {
  u32x4 g0;
  g0[0] = 1u;                                    // count=1, user descriptor
  g0[1] = lds_off;                               // lds_addr (bytes)
  g0[2] = (unsigned)ga;                          // global_addr[31:0]
  g0[3] = (unsigned)((ga >> 32) & 0x01FFFFFFull) | (2u << 30);  // type=2
  i32x8 g1;
  g1[0] = 2 << 16;                               // data_size = 4 bytes
  g1[1] = (int)((unsigned)(tensor_d0 & 0xFFFF) << 16);          // dim0 lo
  g1[2] = (int)((((unsigned)tensor_d0 >> 16) & 0xFFFFu) |
                ((unsigned)(tensor_d1 & 0xFFFF) << 16));
  g1[3] = (int)((((unsigned)tensor_d1 >> 16) & 0xFFFFu) |
                ((unsigned)(tile_d0 & 0xFFFF) << 16));
  g1[4] = tile_d1 & 0xFFFF;                      // tile_dim1 (tile_dim2 = 0)
  g1[5] = stride0;                               // tensor_dim0_stride[31:0]
  g1[6] = 0;                                     // stride hi / dim1_stride lo
  g1[7] = 0;
  i32x4 z4 = {0, 0, 0, 0};                       // groups 2/3 unused (2D)
  i32x8 z8 = {0, 0, 0, 0, 0, 0, 0, 0};
  __builtin_amdgcn_tensor_load_to_lds(g0, g1, z4, z4, z8, 0);
}
__device__ inline void wait_tensor0() {
#if __has_builtin(__builtin_amdgcn_s_wait_tensorcnt)
  __builtin_amdgcn_s_wait_tensorcnt(0);
#else
  asm volatile("s_wait_tensorcnt 0x0" ::: "memory");
#endif
}
#endif  // HAVE_TDM

// ---------------------------------------------------------------------------
// Kernel 1: q = concat(enl, load) @ Wq^T  -> f16 [B*N, 128]
// ---------------------------------------------------------------------------
__global__ __launch_bounds__(32) void qproj_kernel(
    const float* __restrict__ enl, const float* __restrict__ ld,
    const float* __restrict__ Wq, _Float16* __restrict__ qout) {
  const int lane = threadIdx.x;
  const int half = lane >> 4, l16 = lane & 15;
  const int wid = blockIdx.x;
  const int rt = wid >> 3;  // row tile of B*N
  const int ct = wid & 7;   // col tile of 128
  const int rowA = rt * 16 + l16;
  const int colB = ct * 16 + l16;

  v8f acc = zero_v8f();
  const float* ap = enl + (size_t)rowA * cE;
  const float* bp = Wq + (size_t)colB * (cE + 1);
#pragma unroll
  for (int kc = 0; kc < 4; ++kc) {
    const int k0 = kc * 32 + half * 8;
    const int k1 = k0 + 16;
    v16h a, b;
#pragma unroll
    for (int j = 0; j < 8; ++j) {
      a[j]     = (_Float16)ap[k0 + j];
      a[j + 8] = (_Float16)ap[k1 + j];
      b[j]     = (_Float16)bp[k0 + j];
      b[j + 8] = (_Float16)bp[k1 + j];
    }
    acc = wmma_f16(a, b, acc);
  }
  const float wl = bp[cE];  // rank-1 load-scalar column
  const float* lp = ld + rt * 16 + half * 8;
#pragma unroll
  for (int r = 0; r < 8; ++r) {
    const float qv = acc[r] + lp[r] * wl;
    qout[(size_t)(rt * 16 + half * 8 + r) * cHD + colB] = (_Float16)qv;
  }
}

// ---------------------------------------------------------------------------
// Kernel 2: fused dual-branch flash attention, one wave per (b, ntile, head).
// s^T = K*q^T so softmax-over-L is lane-local (one shfl_xor(16)).
// k/v chunks staged into LDS by the Tensor Data Mover (zero-filled tails).
// ---------------------------------------------------------------------------
__device__ inline v8f attn_branch(const v16h qB, const float* __restrict__ kb,
                                  const float* __restrict__ vb,
                                  const float* __restrict__ mrow, const int L,
                                  const int lane, float* klds, float* vlds) {
  const int half = lane >> 4, l16 = lane & 15;
  float m_run = NEG, s_run = 0.0f;
  v8f acc = zero_v8f();
  const int nch = (L + 31) >> 5;
#if HAVE_TDM
  const unsigned koff = (unsigned)(size_t)klds;
  const unsigned voff = (unsigned)(size_t)vlds;
#endif
  for (int c = 0; c < nch; ++c) {
    const int l0 = c * 32;
#if HAVE_TDM
    wait_dscnt0();  // previous chunk's LDS reads done before DMA overwrite
    tdm_load_2d(koff, (unsigned long long)(size_t)(kb + (size_t)l0 * cD),
                cD, L - l0, cD, 32, cD);
    tdm_load_2d(voff, (unsigned long long)(size_t)(vb + (size_t)l0 * cD),
                cD, L - l0, cD, 32, cD);
    wait_tensor0();
#else
    {
      __builtin_prefetch(kb + (size_t)(l0 + 32) * cD, 0, 1);
      __builtin_prefetch(vb + (size_t)(l0 + 32) * cD, 0, 1);
      const int lr = l0 + lane;
      const int lc = lr < L ? lr : L - 1;
      const bool ok = lr < L;
      const float4* kp4 = (const float4*)(kb + (size_t)lc * cD);
      const float4* vp4 = (const float4*)(vb + (size_t)lc * cD);
      const float4 zf4 = make_float4(0.f, 0.f, 0.f, 0.f);
      float4* kd = (float4*)(klds + lane * cD);
      float4* vd = (float4*)(vlds + lane * cD);
#pragma unroll
      for (int i = 0; i < 4; ++i) {
        kd[i] = ok ? kp4[i] : zf4;
        vd[i] = ok ? vp4[i] : zf4;
      }
    }
#endif
    // ---- two 16-L score tiles: s^T = k_tile * qB (D padded 16->32) ----
    v8f s0, s1;
#pragma unroll
    for (int t = 0; t < 2; ++t) {
      const float* kr = klds + (t * 16 + l16) * cD + half * 8;
      v16h a = zero_v16h();
#pragma unroll
      for (int j = 0; j < 8; ++j) a[j] = (_Float16)kr[j];
      const v8f st = wmma_f16(a, qB, zero_v8f());
      if (t == 0) s0 = st; else s1 = st;
    }
    // ---- scale + additive mask + chunk max ----
    float cmax = NEG;
#pragma unroll
    for (int t = 0; t < 2; ++t) {
      v8f& st = t ? s1 : s0;
      const int lb = l0 + t * 16 + half * 8;
#pragma unroll
      for (int r = 0; r < 8; ++r) {
        const int l = lb + r;
        const int lc = l < L ? l : L - 1;
        const float mk = mrow[lc];
        const float sv = (l < L) ? st[r] * INV_SQRT_D + mk : NEG;
        st[r] = sv;
        cmax = fmaxf(cmax, sv);
      }
    }
    cmax = fmaxf(cmax, __shfl_xor(cmax, 16, 32));
    const float m_new = fmaxf(m_run, cmax);
    const float corr = __expf(m_run - m_new);
    s_run *= corr;
#pragma unroll
    for (int r = 0; r < 8; ++r) acc[r] *= corr;
    // ---- p = exp(s - m_new); lane-local repack into B operand ----
    v16h pB;
#pragma unroll
    for (int r = 0; r < 8; ++r) {
      const float p0 = __expf(s0[r] - m_new);
      const float p1 = __expf(s1[r] - m_new);
      s_run += p0 + p1;
      pB[r] = (_Float16)p0;
      pB[8 + r] = (_Float16)p1;
    }
    m_run = m_new;
    // ---- A = v^T (16 D x 32 L) read transposed from LDS ----
    v16h vA;
#pragma unroll
    for (int j = 0; j < 8; ++j) {
      vA[j]     = (_Float16)vlds[(half * 8 + j) * cD + l16];
      vA[8 + j] = (_Float16)vlds[(16 + half * 8 + j) * cD + l16];
    }
    acc = wmma_f16(vA, pB, acc);  // out^T[D,N] accumulate
  }
  float tot = s_run + __shfl_xor(s_run, 16, 32);
  const float inv = tot > 0.0f ? 1.0f / tot : 0.0f;
#pragma unroll
  for (int r = 0; r < 8; ++r) acc[r] *= inv;
  return acc;
}

__global__ __launch_bounds__(32) void attn_kernel(
    const _Float16* __restrict__ q, const float* __restrict__ k,
    const float* __restrict__ v, const float* __restrict__ ks,
    const float* __restrict__ vs, const float* __restrict__ mask1,
    const float* __restrict__ mask2, _Float16* __restrict__ oc) {
  __shared__ float klds[32 * cD];
  __shared__ float vlds[32 * cD];
  const int lane = threadIdx.x, half = lane >> 4, l16 = lane & 15;
  const int wid = blockIdx.x;
  const int h = wid & 7;
  const int nt = (wid >> 3) & 15;
  const int b = wid >> 7;
  const int n = nt * 16 + l16;
  const size_t bn = (size_t)b * cN + n;

  v16h qB = zero_v16h();
  {
    const _Float16* qp = q + bn * cHD + h * cD + half * 8;
#pragma unroll
    for (int j = 0; j < 8; ++j) qB[j] = qp[j];
  }
  const float* kb  = k  + (size_t)(b * cH + h) * cM * cD;
  const float* vb  = v  + (size_t)(b * cH + h) * cM * cD;
  const float* ksb = ks + (size_t)(b * cH + h) * cS * cD;
  const float* vsb = vs + (size_t)(b * cH + h) * cS * cD;
  const float* m1 = mask1 + bn * cM;
  const float* m2 = mask2 + bn * cS;

  const v8f o1 = attn_branch(qB, kb, vb, m1, cM, lane, klds, vlds);
  const v8f o2 = attn_branch(qB, ksb, vsb, m2, cS, lane, klds, vlds);

  h8 outv;
#pragma unroll
  for (int r = 0; r < 8; ++r) outv[r] = (_Float16)(o1[r] + o2[r]);
  *(h8*)(oc + bn * cHD + h * cD + half * 8) = outv;  // contiguous 16B store
}

// ---------------------------------------------------------------------------
// Kernel 3: mh = oc @ Wc^T (LDS-resident), then score^T = SHK_tile * mh^T,
// logits = 10*tanh(score/sqrt(E)) + mask; streaming row max/sum.
// single_head_key tiles staged by TDM (coalesced DMA, zero-filled m-tail).
// ---------------------------------------------------------------------------
__global__ __launch_bounds__(32) void score_kernel(
    const _Float16* __restrict__ oc, const float* __restrict__ Wc,
    const float* __restrict__ shk, const float* __restrict__ mask1,
    float* __restrict__ out, float* __restrict__ rowstats) {
  __shared__ _Float16 mh[16 * cHD];
#if HAVE_TDM
  __shared__ float shkt[cE * 16];  // [e][m_local], 8KB
#endif
  const int lane = threadIdx.x, half = lane >> 4, l16 = lane & 15;
  const int wid = blockIdx.x;
  const int nt = wid & 15;
  const int b = wid >> 4;

  // ---- step 1: mh[16 n][128 e] = oc_tile @ Wc^T ----
  const _Float16* ap_base = oc + (size_t)(b * cN + nt * 16 + l16) * cHD;
  for (int et = 0; et < 8; ++et) {
    const int colE = et * 16 + l16;
    const float* wp = Wc + (size_t)colE * cE;
    v8f acc = zero_v8f();
#pragma unroll
    for (int kc = 0; kc < 4; ++kc) {
      const int k0 = kc * 32 + half * 8;
      const h8 alo = *(const h8*)(ap_base + k0);
      const h8 ahi = *(const h8*)(ap_base + k0 + 16);
      v16h a, bb;
#pragma unroll
      for (int j = 0; j < 8; ++j) {
        a[j] = alo[j];
        a[8 + j] = ahi[j];
        bb[j]     = (_Float16)wp[k0 + j];
        bb[8 + j] = (_Float16)wp[k0 + 16 + j];
      }
      acc = wmma_f16(a, bb, acc);
    }
#pragma unroll
    for (int r = 0; r < 8; ++r)
      mh[(half * 8 + r) * cHD + colE] = (_Float16)acc[r];
  }

  // ---- step 2: logits over M, transposed tiles; lane-local softmax stats ---
  const int n = nt * 16 + l16;
  const size_t bn = (size_t)b * cN + n;
  const float* mrow = mask1 + bn * cM;
  float* orow = out + bn * cM;
  const float* shkb = shk + (size_t)b * cE * cM;
  float m_run = NEG, s_run = 0.0f;

  for (int mt = 0; mt < 63; ++mt) {
    const int m0 = mt * 16;
#if HAVE_TDM
    wait_dscnt0();  // previous tile's LDS reads done before DMA overwrite
    tdm_load_2d((unsigned)(size_t)shkt,
                (unsigned long long)(size_t)(shkb + m0), cM - m0, cE, 16, cE,
                cM);
    wait_tensor0();
#endif
    const int mrowA = m0 + l16;
    const int mcl = mrowA < cM ? mrowA : cM - 1;
    (void)mcl;
    v8f acc = zero_v8f();
#pragma unroll
    for (int kc = 0; kc < 4; ++kc) {
      const int e0 = kc * 32 + half * 8;
      v16h a, bb;
#pragma unroll
      for (int j = 0; j < 8; ++j) {
#if HAVE_TDM
        a[j]     = (_Float16)shkt[(e0 + j) * 16 + l16];
        a[8 + j] = (_Float16)shkt[(e0 + 16 + j) * 16 + l16];
#else
        const float x0 = shkb[(size_t)(e0 + j) * cM + mcl];
        const float x1 = shkb[(size_t)(e0 + 16 + j) * cM + mcl];
        a[j]     = mrowA < cM ? (_Float16)x0 : (_Float16)0.0f;
        a[8 + j] = mrowA < cM ? (_Float16)x1 : (_Float16)0.0f;
#endif
      }
      const h8 blo = *(const h8*)(mh + l16 * cHD + e0);
      const h8 bhi = *(const h8*)(mh + l16 * cHD + e0 + 16);
#pragma unroll
      for (int j = 0; j < 8; ++j) {
        bb[j] = blo[j];
        bb[8 + j] = bhi[j];
      }
      acc = wmma_f16(a, bb, acc);
    }
#pragma unroll
    for (int r = 0; r < 8; ++r) {
      const int m = m0 + half * 8 + r;
      const int mc = m < cM ? m : cM - 1;
      float lg = CLIPc * tanhf(acc[r] * INV_SQRT_E) + mrow[mc];
      lg = m < cM ? lg : NEG;
      if (m < cM) orow[m] = lg;
      const float mn = fmaxf(m_run, lg);
      s_run = s_run * __expf(m_run - mn) + __expf(lg - mn);
      m_run = mn;
    }
  }
  const float mo = __shfl_xor(m_run, 16, 32);
  const float so = __shfl_xor(s_run, 16, 32);
  const float Mx = fmaxf(m_run, mo);
  const float Sx = s_run * __expf(m_run - Mx) + so * __expf(mo - Mx);
  if (lane < 16) {
    rowstats[bn * 2] = Mx;
    rowstats[bn * 2 + 1] = Sx;
  }
}

// ---------------------------------------------------------------------------
// Kernel 4: softmax normalize: out = exp(logit - max) / sum
// ---------------------------------------------------------------------------
__global__ void norm_kernel(float* __restrict__ out,
                            const float* __restrict__ rowstats,
                            long long total) {
  long long i = (long long)blockIdx.x * blockDim.x + threadIdx.x;
  const long long stride = (long long)gridDim.x * blockDim.x;
  for (; i < total; i += stride) {
    const long long row = i / cM;
    const float mx = rowstats[row * 2];
    const float sm = rowstats[row * 2 + 1];
    const float p = __expf(out[i] - mx);
    out[i] = sm > 0.0f ? p / sm : 0.0f;
  }
}

extern "C" void kernel_launch(void* const* d_in, const int* in_sizes, int n_in,
                              void* d_out, int out_size, void* d_ws,
                              size_t ws_size, hipStream_t stream) {
  (void)in_sizes; (void)n_in; (void)out_size; (void)ws_size;
  const float* enl = (const float*)d_in[0];
  const float* ld  = (const float*)d_in[1];
  const float* m2  = (const float*)d_in[2];  // sols_mask_pomo [B,N,S]
  const float* m1  = (const float*)d_in[3];  // ninf_mask      [B,N,M]
  const float* k   = (const float*)d_in[4];
  const float* v   = (const float*)d_in[5];
  const float* ks  = (const float*)d_in[6];
  const float* vs  = (const float*)d_in[7];
  const float* shk = (const float*)d_in[8];
  const float* Wq  = (const float*)d_in[9];
  const float* Wc  = (const float*)d_in[10];
  float* out = (float*)d_out;

  char* ws = (char*)d_ws;
  _Float16* qbuf  = (_Float16*)ws;                                // 4 MB
  _Float16* ocbuf = (_Float16*)(ws + (size_t)cB * cN * cHD * 2);  // 4 MB
  float* rowstats = (float*)(ws + (size_t)cB * cN * cHD * 4);     // 128 KB

  hipLaunchKernelGGL(qproj_kernel, dim3((cB * cN / 16) * 8), dim3(32), 0,
                     stream, enl, ld, Wq, qbuf);
  hipLaunchKernelGGL(attn_kernel, dim3(cB * (cN / 16) * cH), dim3(32), 0,
                     stream, qbuf, k, v, ks, vs, m1, m2, ocbuf);
  hipLaunchKernelGGL(score_kernel, dim3(cB * (cN / 16)), dim3(32), 0, stream,
                     ocbuf, Wc, shk, m1, out, rowstats);
  const long long total = (long long)cB * cN * cM;
  hipLaunchKernelGGL(norm_kernel, dim3(2048), dim3(256), 0, stream, out,
                     rowstats, total);
}
